// TangentSpaceLoss_33732673143655
// MI455X (gfx1250) — compile-verified, matrix operand order checked by reading, not run on hardware
//
#include <hip/hip_runtime.h>
#include <hip/hip_bf16.h>

typedef __attribute__((ext_vector_type(16))) __bf16 v16bf;
typedef __attribute__((ext_vector_type(8)))  float  v8f;
typedef __attribute__((ext_vector_type(4)))  unsigned int v4u;
typedef __attribute__((ext_vector_type(4)))  int v4i;
typedef __attribute__((ext_vector_type(8)))  int v8i;

constexpr int Bn = 8192;   // cells
constexpr int Gn = 512;    // genes
constexpr int Dn = 64;     // latent dim
constexpr int JP = 4;      // j-range partitions
constexpr int NTILE = Bn / 16;       // 512 j-tiles
constexpr int TPP = NTILE / JP;      // 128 tiles per partition

// LDS tile geometry (bf16, TDM row padding keeps 16B alignment + bank spread)
constexpr int VROW = 1040;           // 512*2 data + 16 pad  (260 dwords)
constexpr int EROW = 144;            // 64*2 data + 16 pad   (36 dwords)

// workspace layout (bytes)
constexpr size_t O_SQN = 0;                 // 8192 f32  ||v_i||^2
constexpr size_t O_M   = 32768;             // 8192 f32  softmax max bound
constexpr size_t O_MAX = 65536;             // 1 f32     max ||e_j||^2
constexpr size_t O_DEN = 69632;             // 8192 f32
constexpr size_t O_NUM = 102400;            // 8192 f32
constexpr size_t O_VBF = 135168;            // 8192*512 bf16 (8 MB)
constexpr size_t O_EBF = O_VBF + (size_t)Bn * Gn * 2;   // 8192*64 bf16 (1 MB)
constexpr size_t WS_NEED = O_EBF + (size_t)Bn * Dn * 2;

__device__ __forceinline__ unsigned short f2bf(float f) {
    unsigned int u = __float_as_uint(f);
    u += 0x7fffu + ((u >> 16) & 1u);
    return (unsigned short)(u >> 16);
}

// ---------------- init kernel ----------------
__global__ __launch_bounds__(256) void zero_kernel(float* out, float* mx,
                                                   float* den, float* num, int big) {
    const int i = blockIdx.x * blockDim.x + threadIdx.x;
    if (big && i < Bn) { den[i] = 0.0f; num[i] = 0.0f; }
    if (i == 0) { out[0] = 0.0f; if (big) mx[0] = 0.0f; }
}

// ---------------- row norms of V (and E + global max when big) ----------------
__global__ __launch_bounds__(256) void rownorm_kernel(const float* __restrict__ V,
                                                      const float* __restrict__ E,
                                                      float* __restrict__ sqn,
                                                      float* __restrict__ M,
                                                      float* __restrict__ mx,
                                                      int write_m) {
    const int row  = (blockIdx.x * blockDim.x + threadIdx.x) >> 5;
    const int lane = threadIdx.x & 31;
    if (row >= Bn) return;
    const float* rv = V + (size_t)row * Gn;
    float sv = 0.0f;
    #pragma unroll
    for (int k = lane; k < Gn; k += 32) { float x = rv[k]; sv = fmaf(x, x, sv); }
    float se = 0.0f;
    if (write_m) {
        const float* re = E + (size_t)row * Dn;
        #pragma unroll
        for (int k = lane; k < Dn; k += 32) { float x = re[k]; se = fmaf(x, x, se); }
    }
    #pragma unroll
    for (int msk = 1; msk < 32; msk <<= 1) {
        sv += __shfl_xor(sv, msk);
        se += __shfl_xor(se, msk);
    }
    if (lane == 0) {
        sqn[row] = sv;
        if (write_m) {
            M[row] = se;                                     // raw ||e_row||^2
            atomicMax((int*)mx, __float_as_int(se));         // positive floats: int order ok
        }
    }
}

// ---------------- f32 -> packed bf16 pre-conversion ----------------
__global__ __launch_bounds__(256) void preconv_kernel(const float2* __restrict__ src,
                                                      unsigned int* __restrict__ dst, int n2) {
    for (int i = blockIdx.x * blockDim.x + threadIdx.x; i < n2; i += gridDim.x * blockDim.x) {
        const float2 f = src[i];
        dst[i] = (unsigned)f2bf(f.x) | ((unsigned)f2bf(f.y) << 16);
    }
}

// ---------------- finalize softmax max bound: M_i = ||e_i|| * max_j ||e_j|| ----------------
__global__ __launch_bounds__(256) void mfinal_kernel(float* __restrict__ M,
                                                     const float* __restrict__ mx) {
    const int i = blockIdx.x * blockDim.x + threadIdx.x;
    if (i < Bn) M[i] = sqrtf(M[i]) * sqrtf(mx[0]);
}

// ---------------- TDM issue helper (2D tile descriptor, groups 2/3 unused) ----------------
__device__ __forceinline__ void tdm_issue(unsigned long long ga, unsigned lds_off, v8i g1) {
    v4u g0;
    g0.x = 1u;                                   // count=1, user descriptor
    g0.y = lds_off;                              // lds_addr
    g0.z = (unsigned)ga;                         // global_addr[31:0]
    g0.w = ((unsigned)(ga >> 32) & 0x1FFFFFFu)   // global_addr[56:32]
         | 0x80000000u;                          // type=2 ("image")
    const v4i z4 = {0, 0, 0, 0};
    const v8i z8 = {0, 0, 0, 0, 0, 0, 0, 0};
    __builtin_amdgcn_tensor_load_to_lds(g0, g1, z4, z4, z8, 0);
}

// ---------------- main fused kernel: TDM double-buffered, fixed-max softmax ----------------
__global__ __launch_bounds__(256, 1)
void tangent_tdm_kernel(const unsigned short* __restrict__ Vbf,
                        const unsigned short* __restrict__ Ebf,
                        const float* __restrict__ sqn,
                        const float* __restrict__ Mrow,
                        float* __restrict__ denA,
                        float* __restrict__ numA) {
    __shared__ alignas(16) unsigned char s_v[2][16 * VROW];
    __shared__ alignas(16) unsigned char s_e[2][16 * EROW];

    const int tid  = threadIdx.x;
    const int lane = tid & 31;
    const int wid  = tid >> 5;
    const int half = (lane >> 4) & 1;
    const int m    = lane & 15;
    const int ib   = blockIdx.x & 63;         // i-block
    const int part = blockIdx.x >> 6;         // j partition
    const int i0   = ib * 128 + wid * 16;
    const int jt0  = part * TPP;

    int k0p[8];
    #pragma unroll
    for (int p = 0; p < 8; ++p)
        k0p[p] = (p < 4 ? 2 * p : 16 + 2 * (p - 4)) + 8 * half;

    // resident bf16 A operands from pre-converted global copies
    v16bf a_g[16];
    v16bf a_s[2];
    {
        const unsigned short* rowV = Vbf + (size_t)(i0 + m) * Gn;
        #pragma unroll
        for (int kc = 0; kc < 16; ++kc) {
            union { v16bf v; unsigned int u[8]; } t;
            #pragma unroll
            for (int p = 0; p < 8; ++p)
                t.u[p] = *(const unsigned int*)(rowV + k0p[p] + 32 * kc);
            a_g[kc] = t.v;
        }
        const unsigned short* rowE = Ebf + (size_t)(i0 + m) * Dn;
        #pragma unroll
        for (int kc = 0; kc < 2; ++kc) {
            union { v16bf v; unsigned int u[8]; } t;
            #pragma unroll
            for (int p = 0; p < 8; ++p)
                t.u[p] = *(const unsigned int*)(rowE + k0p[p] + 32 * kc);
            a_s[kc] = t.v;
        }
    }

    float sqn_i[8], Mi[8];
    #pragma unroll
    for (int r = 0; r < 8; ++r) {
        sqn_i[r] = sqn[i0 + r + 8 * half];
        Mi[r]    = Mrow[i0 + r + 8 * half];
    }

    float den[8], num[8];
    #pragma unroll
    for (int r = 0; r < 8; ++r) { den[r] = 0.0f; num[r] = 0.0f; }

    // TDM descriptors (group1): data_size=4B, pad_enable, pad = 4 dwords
    // V: row = 256 dwords (interval code 7), tensor 256x8192, tile 256x16, stride 256
    const v8i VG1 = {0x07D20000, 0x01000000, 0x20000000, 0x01000000, 16, 256, 0, 0};
    // E: row = 32 dwords (interval code 4), tensor 32x8192, tile 32x16, stride 32
    const v8i EG1 = {0x07120000, 0x00200000, 0x20000000, 0x00200000, 16, 32, 0, 0};

    const unsigned vo0 = (unsigned)(uintptr_t)&s_v[0][0];
    const unsigned vo1 = (unsigned)(uintptr_t)&s_v[1][0];
    const unsigned eo0 = (unsigned)(uintptr_t)&s_e[0][0];
    const unsigned eo1 = (unsigned)(uintptr_t)&s_e[1][0];
    const unsigned long long vga = (unsigned long long)(uintptr_t)Vbf;
    const unsigned long long ega = (unsigned long long)(uintptr_t)Ebf;

    // prologue: stage tile 0 into buffer 0
    if (wid == 0) {
        tdm_issue(vga + (unsigned long long)jt0 * (16 * Gn * 2), vo0, VG1);
        tdm_issue(ega + (unsigned long long)jt0 * (16 * Dn * 2), eo0, EG1);
    }

    for (int t = 0; t < TPP; ++t) {
        const int jt  = jt0 + t;
        const int buf = t & 1;

        if (wid == 0) {
            if (t + 1 < TPP) {   // prefetch next tile into the other buffer
                const unsigned long long vj = vga + (unsigned long long)(jt + 1) * (16 * Gn * 2);
                const unsigned long long ej = ega + (unsigned long long)(jt + 1) * (16 * Dn * 2);
                tdm_issue(vj, buf ? vo0 : vo1, VG1);
                tdm_issue(ej, buf ? eo0 : eo1, EG1);
                __builtin_amdgcn_s_wait_tensorcnt(2);   // current tile complete
            } else {
                __builtin_amdgcn_s_wait_tensorcnt(0);
            }
        }
        __syncthreads();

        const float sqn_j = sqn[jt * 16 + m];

        const unsigned char* vb = &s_v[buf][0] + m * VROW + half * 16;
        v8f cg = {0.0f, 0.0f, 0.0f, 0.0f, 0.0f, 0.0f, 0.0f, 0.0f};
        #pragma unroll
        for (int kc = 0; kc < 16; ++kc) {
            const uint4 lo = *(const uint4*)(vb + kc * 64);
            const uint4 hi = *(const uint4*)(vb + kc * 64 + 32);
            union { v16bf v; unsigned int u[8]; } b;
            b.u[0] = lo.x; b.u[1] = lo.y; b.u[2] = lo.z; b.u[3] = lo.w;
            b.u[4] = hi.x; b.u[5] = hi.y; b.u[6] = hi.z; b.u[7] = hi.w;
            cg = __builtin_amdgcn_wmma_f32_16x16x32_bf16(false, a_g[kc], false, b.v,
                                                         (short)0, cg, false, false);
        }

        const unsigned char* eb = &s_e[buf][0] + m * EROW + half * 16;
        v8f cs = {0.0f, 0.0f, 0.0f, 0.0f, 0.0f, 0.0f, 0.0f, 0.0f};
        #pragma unroll
        for (int kc = 0; kc < 2; ++kc) {
            const uint4 lo = *(const uint4*)(eb + kc * 64);
            const uint4 hi = *(const uint4*)(eb + kc * 64 + 32);
            union { v16bf v; unsigned int u[8]; } b;
            b.u[0] = lo.x; b.u[1] = lo.y; b.u[2] = lo.z; b.u[3] = lo.w;
            b.u[4] = hi.x; b.u[5] = hi.y; b.u[6] = hi.z; b.u[7] = hi.w;
            cs = __builtin_amdgcn_wmma_f32_16x16x32_bf16(false, a_s[kc], false, b.v,
                                                         (short)0, cs, false, false);
        }

        // fixed-max softmax accumulation: no cross-lane ops in the loop
        #pragma unroll
        for (int r = 0; r < 8; ++r) {
            const float w = sqn_i[r] + sqn_j - 2.0f * cg[r];
            const float e = __expf(cs[r] - Mi[r]);
            den[r] += e;
            num[r]  = fmaf(e, w, num[r]);
        }
        __syncthreads();
    }

    // one-time reduction across the 16-lane halves, then per-row atomics
    #pragma unroll
    for (int r = 0; r < 8; ++r) {
        #pragma unroll
        for (int msk = 1; msk < 16; msk <<= 1) {
            den[r] += __shfl_xor(den[r], msk);
            num[r] += __shfl_xor(num[r], msk);
        }
    }
    if (m == 0) {
        #pragma unroll
        for (int r = 0; r < 8; ++r) {
            const int row = i0 + r + 8 * half;
            unsafeAtomicAdd(&denA[row], den[r]);
            unsafeAtomicAdd(&numA[row], num[r]);
        }
    }
}

// ---------------- final: loss = (1/B^2) * sum_i num_i/den_i ----------------
__global__ __launch_bounds__(256) void finalsum_kernel(const float* __restrict__ denA,
                                                       const float* __restrict__ numA,
                                                       float* __restrict__ out) {
    const int i    = blockIdx.x * blockDim.x + threadIdx.x;
    const int lane = threadIdx.x & 31;
    float t = (i < Bn) ? (numA[i] / denA[i]) : 0.0f;
    #pragma unroll
    for (int msk = 1; msk < 32; msk <<= 1) t += __shfl_xor(t, msk);
    if (lane == 0) {
        const float inv = 1.0f / ((float)Bn * (float)Bn);
        unsafeAtomicAdd(out, t * inv);
    }
}

// ---------------- fallback (small ws): round-1 style fused kernel ----------------
__global__ __launch_bounds__(256, 1)
void tangent_fallback_kernel(const float* __restrict__ V,
                             const float* __restrict__ E,
                             const float* __restrict__ sqn,
                             float* __restrict__ out) {
    constexpr int LDV = 520;
    constexpr int LDE = 72;
    __shared__ unsigned short Vj[16 * LDV];
    __shared__ unsigned short Ej[16 * LDE];

    const int tid  = threadIdx.x;
    const int lane = tid & 31;
    const int wid  = tid >> 5;
    const int half = (lane >> 4) & 1;
    const int m    = lane & 15;
    const int i0   = blockIdx.x * 128 + wid * 16;

    int k0p[8];
    #pragma unroll
    for (int p = 0; p < 8; ++p)
        k0p[p] = (p < 4 ? 2 * p : 16 + 2 * (p - 4)) + 8 * half;

    v16bf a_g[16]; v16bf a_s[2];
    {
        const float* rowV = V + (size_t)(i0 + m) * Gn;
        #pragma unroll
        for (int kc = 0; kc < 16; ++kc) {
            union { v16bf v; unsigned short s[16]; } t;
            #pragma unroll
            for (int p = 0; p < 8; ++p) {
                const int k = k0p[p] + 32 * kc;
                t.s[2 * p] = f2bf(rowV[k]); t.s[2 * p + 1] = f2bf(rowV[k + 1]);
            }
            a_g[kc] = t.v;
        }
        const float* rowE = E + (size_t)(i0 + m) * Dn;
        #pragma unroll
        for (int kc = 0; kc < 2; ++kc) {
            union { v16bf v; unsigned short s[16]; } t;
            #pragma unroll
            for (int p = 0; p < 8; ++p) {
                const int k = k0p[p] + 32 * kc;
                t.s[2 * p] = f2bf(rowE[k]); t.s[2 * p + 1] = f2bf(rowE[k + 1]);
            }
            a_s[kc] = t.v;
        }
    }

    float sqn_i[8];
    #pragma unroll
    for (int r = 0; r < 8; ++r) sqn_i[r] = sqn[i0 + r + 8 * half];

    float run_max[8], den[8], num[8];
    #pragma unroll
    for (int r = 0; r < 8; ++r) { run_max[r] = -__builtin_inff(); den[r] = 0.0f; num[r] = 0.0f; }

    for (int jt = 0; jt < NTILE; ++jt) {
        const int j0 = jt * 16;
        #pragma unroll 4
        for (int idx = tid; idx < 2048; idx += 256) {
            const int row = idx >> 7, c4 = idx & 127;
            const float4 f = *(const float4*)(V + (size_t)(j0 + row) * Gn + c4 * 4);
            *(unsigned int*)&Vj[row * LDV + c4 * 4]     = (unsigned)f2bf(f.x) | ((unsigned)f2bf(f.y) << 16);
            *(unsigned int*)&Vj[row * LDV + c4 * 4 + 2] = (unsigned)f2bf(f.z) | ((unsigned)f2bf(f.w) << 16);
        }
        {
            const int row = tid >> 4, c4 = tid & 15;
            const float4 f = *(const float4*)(E + (size_t)(j0 + row) * Dn + c4 * 4);
            *(unsigned int*)&Ej[row * LDE + c4 * 4]     = (unsigned)f2bf(f.x) | ((unsigned)f2bf(f.y) << 16);
            *(unsigned int*)&Ej[row * LDE + c4 * 4 + 2] = (unsigned)f2bf(f.z) | ((unsigned)f2bf(f.w) << 16);
        }
        __syncthreads();

        const float sqn_j = sqn[j0 + m];
        v8f cg = {0.0f, 0.0f, 0.0f, 0.0f, 0.0f, 0.0f, 0.0f, 0.0f};
        #pragma unroll
        for (int kc = 0; kc < 16; ++kc) {
            union { v16bf v; unsigned int u[8]; } b;
            #pragma unroll
            for (int p = 0; p < 8; ++p)
                b.u[p] = *(const unsigned int*)&Vj[m * LDV + k0p[p] + 32 * kc];
            cg = __builtin_amdgcn_wmma_f32_16x16x32_bf16(false, a_g[kc], false, b.v,
                                                         (short)0, cg, false, false);
        }
        v8f cs = {0.0f, 0.0f, 0.0f, 0.0f, 0.0f, 0.0f, 0.0f, 0.0f};
        #pragma unroll
        for (int kc = 0; kc < 2; ++kc) {
            union { v16bf v; unsigned int u[8]; } b;
            #pragma unroll
            for (int p = 0; p < 8; ++p)
                b.u[p] = *(const unsigned int*)&Ej[m * LDE + k0p[p] + 32 * kc];
            cs = __builtin_amdgcn_wmma_f32_16x16x32_bf16(false, a_s[kc], false, b.v,
                                                         (short)0, cs, false, false);
        }
        #pragma unroll
        for (int r = 0; r < 8; ++r) {
            const float l = cs[r];
            const float w = sqn_i[r] + sqn_j - 2.0f * cg[r];
            float tm = l;
            tm = fmaxf(tm, __shfl_xor(tm, 1)); tm = fmaxf(tm, __shfl_xor(tm, 2));
            tm = fmaxf(tm, __shfl_xor(tm, 4)); tm = fmaxf(tm, __shfl_xor(tm, 8));
            const float nm = fmaxf(run_max[r], tm);
            const float sc = __expf(run_max[r] - nm);
            float e  = __expf(l - nm);
            float ew = e * w;
            e += __shfl_xor(e, 1);  ew += __shfl_xor(ew, 1);
            e += __shfl_xor(e, 2);  ew += __shfl_xor(ew, 2);
            e += __shfl_xor(e, 4);  ew += __shfl_xor(ew, 4);
            e += __shfl_xor(e, 8);  ew += __shfl_xor(ew, 8);
            den[r] = den[r] * sc + e;
            num[r] = num[r] * sc + ew;
            run_max[r] = nm;
        }
        __syncthreads();
    }

    float partial = 0.0f;
    #pragma unroll
    for (int r = 0; r < 8; ++r) partial += num[r] / den[r];
    float contrib = (m == 0) ? partial : 0.0f;
    #pragma unroll
    for (int msk = 1; msk < 32; msk <<= 1) contrib += __shfl_xor(contrib, msk);
    if (lane == 0) unsafeAtomicAdd(out, contrib * (1.0f / ((float)Bn * (float)Bn)));
}

extern "C" void kernel_launch(void* const* d_in, const int* in_sizes, int n_in,
                              void* d_out, int out_size, void* d_ws, size_t ws_size,
                              hipStream_t stream) {
    const float* V = (const float*)d_in[0];
    const float* E = (const float*)d_in[1];
    float* out     = (float*)d_out;
    char*  ws      = (char*)d_ws;

    float* sqn = (float*)(ws + O_SQN);
    float* M   = (float*)(ws + O_M);
    float* mx  = (float*)(ws + O_MAX);
    float* den = (float*)(ws + O_DEN);
    float* num = (float*)(ws + O_NUM);
    unsigned short* Vbf = (unsigned short*)(ws + O_VBF);
    unsigned short* Ebf = (unsigned short*)(ws + O_EBF);

    const int big = (ws_size >= WS_NEED) ? 1 : 0;

    zero_kernel<<<32, 256, 0, stream>>>(out, mx, den, num, big);
    rownorm_kernel<<<1024, 256, 0, stream>>>(V, E, sqn, M, mx, big);

    if (big) {
        preconv_kernel<<<1024, 256, 0, stream>>>((const float2*)V, (unsigned int*)Vbf,
                                                 Bn * Gn / 2);
        preconv_kernel<<<256, 256, 0, stream>>>((const float2*)E, (unsigned int*)Ebf,
                                                Bn * Dn / 2);
        mfinal_kernel<<<32, 256, 0, stream>>>(M, mx);
        tangent_tdm_kernel<<<64 * JP, 256, 0, stream>>>(Vbf, Ebf, sqn, M, den, num);
        finalsum_kernel<<<32, 256, 0, stream>>>(den, num, out);
    } else {
        tangent_fallback_kernel<<<64, 256, 0, stream>>>(V, E, sqn, out);
    }
}
